// MultiHeadedAttention_29437705847592
// MI455X (gfx1250) — matrix operand order repaired and checked by reading, not compiled, not run on hardware
//
#include <hip/hip_runtime.h>
#include <hip/hip_bf16.h>

// MHA forward with RoPE for MI455X (gfx1250), bf16 WMMA + fp32 accumulate.
// Shapes: B=2, T=2048, D=1024, H=16, DH=64.
#define D_  1024
#define H_  16
#define DH_ 64
#define T_  2048
#define B_  2
#define BT_ (B_ * T_)

typedef __attribute__((ext_vector_type(16))) __bf16 v16bf;
typedef __attribute__((ext_vector_type(8)))  __bf16 v8bf;
typedef __attribute__((ext_vector_type(8)))  float  v8f;

// A-fragment: 16 bf16 per lane, loaded as two contiguous 8-element chunks
// (K 0..15 for lanes 0-15, 16..31 for lanes 16-31, matching the
// contiguous B-fragment loads).
union AFrag { v16bf v; v8bf h[2]; };

__device__ __forceinline__ v8f wmma_bf16(v16bf a, v16bf b, v8f c) {
  // emits v_wmma_f32_16x16x32_bf16
  return __builtin_amdgcn_wmma_f32_16x16x32_bf16(false, a, false, b, (short)0, c,
                                                 false, false);
}

// Async-stage one 16-byte chunk: global -> LDS, tracked by ASYNCcnt.
__device__ __forceinline__ void async_b128(unsigned lds_off, const void* gaddr) {
  asm volatile("global_load_async_to_lds_b128 %0, %1, off"
               :: "v"(lds_off), "v"(gaddr)
               : "memory");
}

__global__ void cvt_f32_bf16(const float* __restrict__ src,
                             __bf16* __restrict__ dst, int n) {
  int i = blockIdx.x * blockDim.x + threadIdx.x;
  int stride = gridDim.x * blockDim.x;
  for (; i < n; i += stride) dst[i] = (__bf16)src[i];
}

// ---------------------------------------------------------------------------
// QKV projection (C = X * W^T) fused with RoPE on Q/K epilogue.
// Each wave computes a 16(M) x 64(N) output strip: 1 A-frag reused by 4 B-frags
// per 32-wide K step -> 128 WMMA per wave.
// grid = (BT/16, D/256, 3[q,k,v]); block = 128 (4 waves).
// Q,K stored [B,H,T,DH]; V stored transposed [B,H,DH,T] so the later P*V
// B-fragments are contiguous in memory.
// ---------------------------------------------------------------------------
__global__ __launch_bounds__(128) void qkv_rope_gemm(
    const __bf16* __restrict__ Xb, const __bf16* __restrict__ Wqkv,
    __bf16* __restrict__ Q, __bf16* __restrict__ K, __bf16* __restrict__ Vt) {
  const int proj = blockIdx.z;  // 0=Q, 1=K, 2=V
  const __bf16* W = Wqkv + (size_t)proj * D_ * D_;
  const int wave = threadIdx.x >> 5;
  const int lane = threadIdx.x & 31;
  const int lh = lane >> 4, ln = lane & 15;
  const int mBase = blockIdx.x * 16;
  const int nBase = (blockIdx.y * 4 + wave) * 64;

  const v8f vz = {0.f, 0.f, 0.f, 0.f, 0.f, 0.f, 0.f, 0.f};
  v8f acc[4] = {vz, vz, vz, vz};

  const __bf16* arow = Xb + (size_t)(mBase + ln) * D_;
  for (int k0 = 0; k0 < D_; k0 += 32) {
    AFrag a;
    a.h[0] = *(const v8bf*)(arow + k0 + lh * 16);
    a.h[1] = *(const v8bf*)(arow + k0 + lh * 16 + 8);
#pragma unroll
    for (int t = 0; t < 4; ++t) {
      // B column n = W row n (K-contiguous): single 32B contiguous frag load.
      const __bf16* bp = W + (size_t)(nBase + t * 16 + ln) * D_ + k0 + lh * 16;
      v16bf bv = *(const v16bf*)bp;
      acc[t] = wmma_bf16(a.v, bv, acc[t]);
    }
  }

  // Epilogue: C layout -> lane ln holds column n, VGPR j holds row j + 8*lh.
#pragma unroll
  for (int t = 0; t < 4; ++t) {
#pragma unroll
    for (int j = 0; j < 8; ++j) {
      float x = acc[t][j];
      int m  = mBase + j + 8 * lh;
      int bb = m >> 11;          // m / T_
      int tp = m & (T_ - 1);     // m % T_
      int n  = nBase + t * 16 + ln;
      int hh = n >> 6, dh = n & 63;
      if (proj == 2) {
        Vt[(((size_t)(bb * H_ + hh)) * DH_ + dh) * T_ + tp] = (__bf16)x;
      } else {
        // RoPE: pair partner lives in the adjacent lane (n ^ 1).
        float partner = __shfl_xor(x, 1, 32);
        int pairIdx = dh >> 1;
        // theta = 10000^(-2*pairIdx/DH) = exp(-ln(10000)*2*pairIdx/DH)
        float theta = __expf((float)pairIdx * (-2.0f / (float)DH_) * 9.210340371976184f);
        float ang = (float)tp * theta;
        float s, c;
        __sincosf(ang, &s, &c);
        float r = ((dh & 1) == 0) ? (c * x - s * partner)
                                  : (s * partner + c * x);
        __bf16* dst = (proj == 0) ? Q : K;
        dst[(((size_t)(bb * H_ + hh)) * T_ + tp) * DH_ + dh] = (__bf16)r;
      }
    }
  }
}

// ---------------------------------------------------------------------------
// Flash attention: grid = (T/64, H, B), block = 128 (4 waves); each wave owns
// 16 query rows. K / V^T tiles are async-staged into double-buffered LDS
// (global_load_async_to_lds_b128, ASYNCcnt) so tile j+1 streams while tile j
// is consumed; all 4 waves share the staged tiles (4x less global traffic).
// Online softmax in fp32; P re-laid-out through LDS (bf16) to feed P*V WMMA.
// ---------------------------------------------------------------------------
__global__ __launch_bounds__(128) void flash_attn(
    const __bf16* __restrict__ Q, const __bf16* __restrict__ K,
    const __bf16* __restrict__ Vt, __bf16* __restrict__ Hb) {
  __shared__ __bf16 Klds[2][64][64];   // [buf][key][dh]
  __shared__ __bf16 Vlds[2][64][64];   // [buf][dh][key]
  __shared__ __bf16 Plds[4][16][64];   // [wave][q][key]

  const int qt = blockIdx.x, hh = blockIdx.y, b = blockIdx.z;
  const int tid = threadIdx.x;
  const int wave = tid >> 5, lane = tid & 31;
  const int lh = lane >> 4, ln = lane & 15;
  const size_t head = ((size_t)(b * H_ + hh)) * T_ * DH_;
  const __bf16* Qh = Q + head;
  const __bf16* Kh = K + head;
  const __bf16* Vh = Vt + head;  // [DH][T]
  const int qBase = qt * 64 + wave * 16;

  // Q fragments over d = 0..63 (two K=32 chunks), loaded once.
  AFrag qf[2];
  {
    const __bf16* row = Qh + (size_t)(qBase + ln) * DH_;
#pragma unroll
    for (int c2 = 0; c2 < 2; ++c2) {
      qf[c2].h[0] = *(const v8bf*)(row + c2 * 32 + lh * 16);
      qf[c2].h[1] = *(const v8bf*)(row + c2 * 32 + lh * 16 + 8);
    }
  }

  // Per-tile staging: 64x64 bf16 K tile + 64x64 bf16 V^T tile = 512+512
  // 16-byte chunks; 128 threads x 4 chunks each => 8 async ops per wave.
  auto stage = [&](int buf, int kBase) {
#pragma unroll
    for (int i = 0; i < 4; ++i) {
      int c   = tid + i * 128;   // chunk 0..511
      int row = c >> 3;          // 0..63
      int col = (c & 7) * 8;     // bf16 column, 16B granulated
      async_b128((unsigned)(uintptr_t)&Klds[buf][row][col],
                 (const void*)(Kh + (size_t)(kBase + row) * DH_ + col));
      async_b128((unsigned)(uintptr_t)&Vlds[buf][row][col],
                 (const void*)(Vh + (size_t)row * T_ + kBase + col));
    }
  };

  const v8f vz = {0.f, 0.f, 0.f, 0.f, 0.f, 0.f, 0.f, 0.f};
  v8f accO[4] = {vz, vz, vz, vz};
  float mrun[8], lrun[8];
#pragma unroll
  for (int j = 0; j < 8; ++j) { mrun[j] = -3.0e38f; lrun[j] = 0.f; }

  stage(0, 0);  // prologue: tile 0 in flight

  for (int jt = 0; jt <= qt; ++jt) {
    const int kBase = jt * 64;
    const int buf = jt & 1;

    // Kick off the next tile into the other buffer (safe: that buffer was
    // last read in the previous iteration, behind the end-of-iteration barrier).
    if (jt < qt) {
      stage(buf ^ 1, kBase + 64);
      // Async loads complete in order: <=8 outstanding => this tile's 8 done.
      asm volatile("s_wait_asynccnt 0x8" ::: "memory");
    } else {
      asm volatile("s_wait_asynccnt 0x0" ::: "memory");
    }
    __syncthreads();  // every thread passed its own asynccnt wait

    // S = Qr * Kr^T from LDS: B column n = key row (dh-contiguous).
    v8f accS[4] = {vz, vz, vz, vz};
#pragma unroll
    for (int t = 0; t < 4; ++t)
#pragma unroll
      for (int c = 0; c < 2; ++c) {
        v16bf kv = *(const v16bf*)&Klds[buf][t * 16 + ln][c * 32 + lh * 16];
        accS[t] = wmma_bf16(qf[c].v, kv, accS[t]);
      }

    const bool diag = (jt == qt);
#pragma unroll
    for (int j = 0; j < 8; ++j) {
      const int q = qBase + j + 8 * lh;
      float sv[4];
      float mx = -3.0e38f;
#pragma unroll
      for (int t = 0; t < 4; ++t) {
        float s = accS[t][j] * 0.125f;  // DH^-0.5
        if (diag && (kBase + t * 16 + ln) > q) s = -3.0e38f;  // causal mask
        sv[t] = s;
        mx = fmaxf(mx, s);
      }
      // Row-wise reduce across the 16 lanes holding this row (masks <16 keep
      // the two wave halves separate, matching the C-matrix layout).
#pragma unroll
      for (int m2 = 1; m2 < 16; m2 <<= 1) mx = fmaxf(mx, __shfl_xor(mx, m2, 32));
      float mnew = fmaxf(mrun[j], mx);
      float alpha = __expf(mrun[j] - mnew);
      mrun[j] = mnew;
      float rs = 0.f;
#pragma unroll
      for (int t = 0; t < 4; ++t) {
        float p = __expf(sv[t] - mnew);
        rs += p;
        Plds[wave][j + 8 * lh][t * 16 + ln] = (__bf16)p;
      }
#pragma unroll
      for (int m2 = 1; m2 < 16; m2 <<= 1) rs += __shfl_xor(rs, m2, 32);
      lrun[j] = lrun[j] * alpha + rs;
#pragma unroll
      for (int t = 0; t < 4; ++t) accO[t][j] = accO[t][j] * alpha;
    }
    __syncthreads();

    // O += P * V : A-frags of P from LDS; B-frags from staged V^T tile.
    AFrag pf[2];
    {
      const __bf16* prow = &Plds[wave][ln][0];
#pragma unroll
      for (int c2 = 0; c2 < 2; ++c2) {
        pf[c2].h[0] = *(const v8bf*)(prow + c2 * 32 + lh * 16);
        pf[c2].h[1] = *(const v8bf*)(prow + c2 * 32 + lh * 16 + 8);
      }
    }
#pragma unroll
    for (int t = 0; t < 4; ++t)
#pragma unroll
      for (int c = 0; c < 2; ++c) {
        v16bf vv = *(const v16bf*)&Vlds[buf][t * 16 + ln][c * 32 + lh * 16];
        accO[t] = wmma_bf16(pf[c].v, vv, accO[t]);
      }
    __syncthreads();  // all reads of this buffer done before it is re-staged
  }

  // O /= l ; store heads in [B,T,D] bf16 for the output projection.
#pragma unroll
  for (int t = 0; t < 4; ++t)
#pragma unroll
    for (int j = 0; j < 8; ++j) {
      float o = accO[t][j] / lrun[j];
      int q = qBase + j + 8 * lh;
      int d = hh * DH_ + t * 16 + ln;
      Hb[((size_t)(b * T_ + q)) * D_ + d] = (__bf16)o;
    }
}

// ---------------------------------------------------------------------------
// Output projection: out = Hb * Wo^T, fp32 stores.
// ---------------------------------------------------------------------------
__global__ __launch_bounds__(128) void out_proj_gemm(
    const __bf16* __restrict__ Hb, const __bf16* __restrict__ Wob,
    float* __restrict__ out) {
  const int wave = threadIdx.x >> 5, lane = threadIdx.x & 31;
  const int lh = lane >> 4, ln = lane & 15;
  const int mBase = blockIdx.x * 16;
  const int nBase = (blockIdx.y * 4 + wave) * 64;

  const v8f vz = {0.f, 0.f, 0.f, 0.f, 0.f, 0.f, 0.f, 0.f};
  v8f acc[4] = {vz, vz, vz, vz};

  const __bf16* arow = Hb + (size_t)(mBase + ln) * D_;
  for (int k0 = 0; k0 < D_; k0 += 32) {
    AFrag a;
    a.h[0] = *(const v8bf*)(arow + k0 + lh * 16);
    a.h[1] = *(const v8bf*)(arow + k0 + lh * 16 + 8);
#pragma unroll
    for (int t = 0; t < 4; ++t) {
      v16bf bv = *(const v16bf*)(Wob + (size_t)(nBase + t * 16 + ln) * D_ +
                                 k0 + lh * 16);
      acc[t] = wmma_bf16(a.v, bv, acc[t]);
    }
  }
#pragma unroll
  for (int t = 0; t < 4; ++t)
#pragma unroll
    for (int j = 0; j < 8; ++j) {
      int m = mBase + j + 8 * lh;
      int n = nBase + t * 16 + ln;
      out[(size_t)m * D_ + n] = acc[t][j];
    }
}

extern "C" void kernel_launch(void* const* d_in, const int* in_sizes, int n_in,
                              void* d_out, int out_size, void* d_ws, size_t ws_size,
                              hipStream_t stream) {
  (void)in_sizes; (void)n_in; (void)out_size; (void)ws_size;
  const float* X  = (const float*)d_in[0];
  const float* Wq = (const float*)d_in[1];
  const float* Wk = (const float*)d_in[2];
  const float* Wv = (const float*)d_in[3];
  const float* Wo = (const float*)d_in[4];

  // Workspace layout (bf16 elements), ~50.3 MB total:
  __bf16* ws   = (__bf16*)d_ws;
  __bf16* Xb   = ws;                                   // BT*D
  __bf16* Wqkv = Xb + (size_t)BT_ * D_;                // 3*D*D (q,k,v)
  __bf16* Wob  = Wqkv + (size_t)3 * D_ * D_;           // D*D
  __bf16* Qr   = Wob + (size_t)D_ * D_;                // [B,H,T,DH]
  __bf16* Kr   = Qr + (size_t)B_ * H_ * T_ * DH_;      // [B,H,T,DH]
  __bf16* Vt   = Kr + (size_t)B_ * H_ * T_ * DH_;      // [B,H,DH,T]
  __bf16* Hb   = Vt + (size_t)B_ * H_ * T_ * DH_;      // [B,T,D]

  cvt_f32_bf16<<<1024, 256, 0, stream>>>(X, Xb, BT_ * D_);
  cvt_f32_bf16<<<512, 256, 0, stream>>>(Wq, Wqkv, D_ * D_);
  cvt_f32_bf16<<<512, 256, 0, stream>>>(Wk, Wqkv + (size_t)D_ * D_, D_ * D_);
  cvt_f32_bf16<<<512, 256, 0, stream>>>(Wv, Wqkv + (size_t)2 * D_ * D_, D_ * D_);
  cvt_f32_bf16<<<512, 256, 0, stream>>>(Wo, Wob, D_ * D_);

  qkv_rope_gemm<<<dim3(BT_ / 16, D_ / 256, 3), 128, 0, stream>>>(Xb, Wqkv, Qr, Kr, Vt);
  flash_attn<<<dim3(T_ / 64, H_, B_), 128, 0, stream>>>(Qr, Kr, Vt, Hb);
  out_proj_gemm<<<dim3(BT_ / 16, D_ / 256, 1), 128, 0, stream>>>(Hb, Wob, (float*)d_out);
}
